// Base_Layer_31748398252319
// MI455X (gfx1250) — matrix operand order repaired and checked by reading, not compiled, not run on hardware
//
#include <hip/hip_runtime.h>

typedef _Float16 h16;
typedef __attribute__((ext_vector_type(16))) _Float16 v16h;
typedef __attribute__((ext_vector_type(8)))  float    v8f;

#define B_SZ   4096
#define T_SZ   200
#define T_PAD  208      // 13 tiles of 16
#define MTILES 13
#define E_SZ   64
#define K1     128      // 2E
#define H1_SZ  80
#define H2_SZ  40
#define K2P    96       // H1 padded to 3 K-chunks of 32
#define N2P    48       // H2 padded to 3 N-tiles of 16
#define KT1    4        // 128/32
#define NT1    5        // 80/16
#define KT2    3        // 96/32
#define NT2    3        // 48/16
#define NTHREADS 256
#define NWAVES   8
#define MASK_VAL (-4294967295.0f)   // -(2^32)+1, matches reference padding

// single v_rcp_f32 instead of the IEEE divide sequence; ~1ulp, far below the
// f16 quantization error already present in the WMMA inputs
__device__ __forceinline__ float fast_sigmoid(float x) {
    return __builtin_amdgcn_rcpf(1.0f + __expf(-x));
}

__global__ __launch_bounds__(NTHREADS)
void din_attention_kernel(const float* __restrict__ queries,
                          const float* __restrict__ keys,
                          const int*   __restrict__ keys_length,
                          const float* __restrict__ W1,
                          const float* __restrict__ b1,
                          const float* __restrict__ W2,
                          const float* __restrict__ b2,
                          const float* __restrict__ kernw,
                          const float* __restrict__ biasp,
                          float* __restrict__ out)
{
    // ---- LDS ----
    alignas(8)  __shared__ h16 sK[T_PAD][E_SZ];         // keys, f16
    __shared__ h16 sQ[K1];                              // only first 64 used
    alignas(32) __shared__ h16 sW1[KT1 * NT1 * 512];    // B-fragment-swizzled
    alignas(32) __shared__ h16 sW2[KT2 * NT2 * 512];    // B-fragment-swizzled (zero padded)
    __shared__ h16 sH1[NWAVES][16 * K2P];               // per-wave sigmoid(h1), K-padded
    __shared__ float sScore[T_PAD];
    __shared__ float sWt[T_PAD];
    __shared__ float sKern[N2P];
    __shared__ float sB1[NT1 * 16];
    __shared__ float sB2[N2P];
    __shared__ float sPool[4][E_SZ];
    __shared__ float sRed[2];

    const int tid  = threadIdx.x;
    const int b    = blockIdx.x;
    const int wave = tid >> 5;
    const int lane = tid & 31;
    const int laneHalf = lane >> 4;   // 0: lanes 0-15, 1: lanes 16-31
    const int lane15   = lane & 15;

    const int klen = keys_length[b];
    const float biasv = biasp[0];

    // ---- stage inputs into LDS ----
    if (tid < E_SZ) sQ[tid] = (h16)queries[(size_t)b * E_SZ + tid];

    // keys: vectorized 16B global loads, packed 8B LDS stores (dominant HBM path)
    {
        const float4* kg = (const float4*)(keys + (size_t)b * T_SZ * E_SZ);
        for (int i = tid; i < (T_SZ * E_SZ) / 4; i += NTHREADS) {
            float4 v = kg[i];
            h16* p = &sK[0][0] + i * 4;
            p[0] = (h16)v.x; p[1] = (h16)v.y; p[2] = (h16)v.z; p[3] = (h16)v.w;
        }
        for (int i = tid; i < (T_PAD - T_SZ) * E_SZ; i += NTHREADS)
            sK[T_SZ + (i >> 6)][i & 63] = (h16)0.0f;
    }

    // W1 -> B-fragment order: frag(kt,nt), per-lane 16 contiguous halves
    for (int idx = tid; idx < KT1 * NT1 * 512; idx += NTHREADS) {
        int frag = idx >> 9, w = idx & 511;
        int l = w >> 4, h = w & 15;
        int kt = frag / NT1, nt = frag % NT1;
        int k = kt * 32 + ((h >> 3) << 4) + ((l >> 4) << 3) + (h & 7);   // 0..127
        int n = nt * 16 + (l & 15);              // 0..79
        sW1[idx] = (h16)W1[k * H1_SZ + n];
    }
    // W2 -> B-fragment order, zero-padded to 96x48
    for (int idx = tid; idx < KT2 * NT2 * 512; idx += NTHREADS) {
        int frag = idx >> 9, w = idx & 511;
        int l = w >> 4, h = w & 15;
        int kt = frag / NT2, nt = frag % NT2;
        int k = kt * 32 + ((h >> 3) << 4) + ((l >> 4) << 3) + (h & 7);
        int n = nt * 16 + (l & 15);
        sW2[idx] = (k < H1_SZ && n < H2_SZ) ? (h16)W2[k * H2_SZ + n] : (h16)0.0f;
    }
    if (tid < NT1 * 16) sB1[tid] = b1[tid];
    if (tid < N2P) {
        sB2[tid]   = (tid < H2_SZ) ? b2[tid]    : 0.0f;
        sKern[tid] = (tid < H2_SZ) ? kernw[tid] : 0.0f;
    }
    // zero h1 scratch (covers the K 80..95 pad columns)
    for (int i = tid; i < NWAVES * 16 * K2P / 2; i += NTHREADS)
        ((unsigned int*)sH1)[i] = 0u;

    __syncthreads();

    // ---- per-wave MLP over 16-row tiles ----
    h16* myH1 = &sH1[wave][0];
    for (int tile = wave; tile < MTILES; tile += NWAVES) {
        const int row0 = tile * 16;

        // GEMM1: (16 x 128) x (128 x 80)
        v8f acc1[NT1] = {};
        #pragma unroll
        for (int kt = 0; kt < KT1; ++kt) {
            v16h a;
            if (kt < 2) {
                // query half of the concat: identical rows (broadcast, hoisted)
                #pragma unroll
                for (int j = 0; j < 8; ++j) {
                    int kb = (j < 4) ? 2 * j : 16 + 2 * (j - 4);
                    int k = kt * 32 + kb + laneHalf * 8;
                    a[2 * j]     = sQ[k];
                    a[2 * j + 1] = sQ[k + 1];
                }
            } else {
                const int m = row0 + lane15;   // A: M = lane%16 in both halves
                #pragma unroll
                for (int j = 0; j < 8; ++j) {
                    int kb = (j < 4) ? 2 * j : 16 + 2 * (j - 4);
                    int k = (kt - 2) * 32 + kb + laneHalf * 8;
                    a[2 * j]     = sK[m][k];
                    a[2 * j + 1] = sK[m][k + 1];
                }
            }
            #pragma unroll
            for (int nt = 0; nt < NT1; ++nt) {
                v16h bf = *(const v16h*)&sW1[((kt * NT1 + nt) << 9) + (lane << 4)];
                acc1[nt] = __builtin_amdgcn_wmma_f32_16x16x32_f16(
                    false, a, false, bf, (short)0, acc1[nt], false, false);
            }
        }

        // bias + sigmoid; C layout -> row-major wave scratch
        #pragma unroll
        for (int nt = 0; nt < NT1; ++nt) {
            #pragma unroll
            for (int r = 0; r < 8; ++r) {
                float v = acc1[nt][r] + sB1[nt * 16 + lane15];
                float s = fast_sigmoid(v);
                int m = r + 8 * laneHalf;
                int n = nt * 16 + lane15;
                myH1[m * K2P + n] = (h16)s;
            }
        }

        // GEMM2: (16 x 96) x (96 x 48)
        v8f acc2[NT2] = {};
        #pragma unroll
        for (int kt = 0; kt < KT2; ++kt) {
            v16h a;
            const h16* rp = &myH1[lane15 * K2P];
            #pragma unroll
            for (int j = 0; j < 8; ++j) {
                int kb = (j < 4) ? 2 * j : 16 + 2 * (j - 4);
                int k = kt * 32 + kb + laneHalf * 8;
                a[2 * j]     = rp[k];
                a[2 * j + 1] = rp[k + 1];
            }
            #pragma unroll
            for (int nt = 0; nt < NT2; ++nt) {
                v16h bf = *(const v16h*)&sW2[((kt * NT2 + nt) << 9) + (lane << 4)];
                acc2[nt] = __builtin_amdgcn_wmma_f32_16x16x32_f16(
                    false, a, false, bf, (short)0, acc2[nt], false, false);
            }
        }

        // score = sigmoid(h2) . kernel  (N across lanes -> butterfly reduce per half)
        #pragma unroll
        for (int r = 0; r < 8; ++r) {
            float p = 0.0f;
            #pragma unroll
            for (int nt = 0; nt < NT2; ++nt) {
                float v = acc2[nt][r] + sB2[nt * 16 + lane15];
                p += fast_sigmoid(v) * sKern[nt * 16 + lane15];
            }
            p += __shfl_xor(p, 1);
            p += __shfl_xor(p, 2);
            p += __shfl_xor(p, 4);
            p += __shfl_xor(p, 8);
            if (lane15 == 0) {
                int t = row0 + r + 8 * laneHalf;
                float sc = p + biasv;
                sScore[t] = (t < klen) ? sc
                          : (t < T_SZ) ? MASK_VAL      // masked, matches reference
                                       : -INFINITY;    // pad rows vanish in softmax
            }
        }
    }
    __syncthreads();

    // ---- softmax over T (wave 0) ----
    if (wave == 0) {
        float mx = -INFINITY;
        for (int t = lane; t < T_PAD; t += 32) mx = fmaxf(mx, sScore[t]);
        #pragma unroll
        for (int o = 16; o > 0; o >>= 1) mx = fmaxf(mx, __shfl_xor(mx, o));
        float sum = 0.0f;
        for (int t = lane; t < T_PAD; t += 32) sum += __expf(sScore[t] - mx);
        #pragma unroll
        for (int o = 16; o > 0; o >>= 1) sum += __shfl_xor(sum, o);
        if (lane == 0) { sRed[0] = mx; sRed[1] = sum; }
    }
    __syncthreads();
    {
        float mx = sRed[0], inv = __builtin_amdgcn_rcpf(sRed[1]);
        for (int t = tid; t < T_PAD; t += NTHREADS)
            sWt[t] = __expf(sScore[t] - mx) * inv;
    }
    __syncthreads();

    // ---- weighted pooling: out[e] = sum_t w[t] * keys[t][e] ----
    {
        const int e = tid & 63;
        const int c = tid >> 6;               // 4 chunks of 52 rows
        int t0 = c * 52;
        int t1 = t0 + 52; if (t1 > T_SZ) t1 = T_SZ;
        float acc = 0.0f;
        for (int t = t0; t < t1; ++t)
            acc += sWt[t] * (float)sK[t][e];
        sPool[c][e] = acc;
    }
    __syncthreads();
    if (tid < E_SZ) {
        float o = sPool[0][tid] + sPool[1][tid] + sPool[2][tid] + sPool[3][tid];
        out[(size_t)b * E_SZ + tid] = o;
    }
}

extern "C" void kernel_launch(void* const* d_in, const int* in_sizes, int n_in,
                              void* d_out, int out_size, void* d_ws, size_t ws_size,
                              hipStream_t stream) {
    (void)in_sizes; (void)n_in; (void)out_size; (void)d_ws; (void)ws_size;
    const float* queries     = (const float*)d_in[0];
    const float* keys        = (const float*)d_in[1];
    const int*   keys_length = (const int*)  d_in[2];
    const float* W1          = (const float*)d_in[3];
    const float* b1          = (const float*)d_in[4];
    const float* W2          = (const float*)d_in[5];
    const float* b2          = (const float*)d_in[6];
    const float* kernw       = (const float*)d_in[7];
    const float* biasp       = (const float*)d_in[8];
    float* out = (float*)d_out;

    din_attention_kernel<<<dim3(B_SZ), dim3(NTHREADS), 0, stream>>>(
        queries, keys, keys_length, W1, b1, W2, b2, kernw, biasp, out);
}
